// MiMoV2Attention_7292854468631
// MI455X (gfx1250) — matrix operand order repaired and checked by reading, not compiled, not run on hardware
//
#include <hip/hip_runtime.h>

#define T_    4096
#define HID_  2048
#define HQ_   32
#define HKV_  8
#define G_    4
#define D_    128
#define DV_   96
#define W_    512
#define QKV_N 5888   // HQ*D + HKV*D + HKV*DV
#define ATT_N 3072   // HQ*DV

typedef __attribute__((ext_vector_type(16))) __bf16       v16bf;
typedef __attribute__((ext_vector_type(8)))  float        v8f;
typedef __attribute__((ext_vector_type(2)))  float        v2f;
typedef __attribute__((ext_vector_type(4)))  float        f32x4;
typedef __attribute__((ext_vector_type(4)))  unsigned int u32x4;
typedef __attribute__((ext_vector_type(2)))  unsigned int u32x2;

__device__ __forceinline__ unsigned short f2bf(float f) {
    unsigned u = __float_as_uint(f);
    return (unsigned short)((u + 0x7fffu + ((u >> 16) & 1u)) >> 16);
}
__device__ __forceinline__ float bf2f(unsigned short h) {
    return __uint_as_float(((unsigned)h) << 16);
}
// fp32 -> (bf16 hi, bf16 lo) split, 4 elements packed into two dwords each
__device__ __forceinline__ void pack_hi_lo(f32x4 a, u32x2& hi, u32x2& lo) {
    unsigned short h0 = f2bf(a.x), h1 = f2bf(a.y), h2 = f2bf(a.z), h3 = f2bf(a.w);
    unsigned short l0 = f2bf(a.x - bf2f(h0)), l1 = f2bf(a.y - bf2f(h1));
    unsigned short l2 = f2bf(a.z - bf2f(h2)), l3 = f2bf(a.w - bf2f(h3));
    hi.x = (unsigned)h0 | ((unsigned)h1 << 16); hi.y = (unsigned)h2 | ((unsigned)h3 << 16);
    lo.x = (unsigned)l0 | ((unsigned)l1 << 16); lo.y = (unsigned)l2 | ((unsigned)l3 << 16);
}

union BfFrag { u32x4 q[2]; v16bf v; };

// A-fragment (16x32 bf16): lane<16 holds K 0..7 & 16..23, lane>=16 holds K 8..15 & 24..31
__device__ __forceinline__ v16bf frag_a(const unsigned short* row, int lg) {
    BfFrag u;
    u.q[0] = *(const u32x4*)(row + lg * 8);
    u.q[1] = *(const u32x4*)(row + lg * 8 + 16);
    return u.v;
}
// B-fragment (32x16 bf16): lane<16 holds K 0..7 & 16..23, lane>=16 holds K 8..15 & 24..31
__device__ __forceinline__ v16bf frag_b(const unsigned short* row, int lg) {
    BfFrag u;
    u.q[0] = *(const u32x4*)(row + lg * 8);
    u.q[1] = *(const u32x4*)(row + lg * 8 + 16);
    return u.v;
}

// ---------------------------------------------------------------------------
// C[M,N] = A[M,K] * B[N,K]^T, fp32 in/out, bf16 hi/lo split WMMA accumulation.
// Tile: 128(M) x 64(N) x 32(K), 256 threads = 8 wave32 waves, double-buffered LDS.
// ---------------------------------------------------------------------------
__global__ __launch_bounds__(256) void gemm_f32_bf16split(
    const float* __restrict__ A, int lda,
    const float* __restrict__ B, int ldb,
    float* __restrict__ C, int ldc, int K)
{
    __shared__ unsigned short sAhi[2][128 * 40];
    __shared__ unsigned short sAlo[2][128 * 40];
    __shared__ unsigned short sBhi[2][64 * 40];
    __shared__ unsigned short sBlo[2][64 * 40];

    const int tid  = threadIdx.x;
    const int wave = tid >> 5;
    const int lane = tid & 31;
    const int lg   = lane >> 4;
    const int lr   = lane & 15;
    const int m0   = blockIdx.y * 128;
    const int n0   = blockIdx.x * 64;

    const v8f vzero = {0.f, 0.f, 0.f, 0.f, 0.f, 0.f, 0.f, 0.f};
    v8f acc[4];
    #pragma unroll
    for (int i = 0; i < 4; i++) acc[i] = vzero;

    auto stage = [&](int buf, int k0) {
        #pragma unroll
        for (int j = 0; j < 4; j++) {           // A tile: 128x32 fp32 = 1024 float4
            int f = tid + 256 * j;
            int r = f >> 3, c = (f & 7) << 2;
            f32x4 a = *(const f32x4*)(A + (size_t)(m0 + r) * lda + (k0 + c));
            u32x2 hi, lo; pack_hi_lo(a, hi, lo);
            *(u32x2*)&sAhi[buf][r * 40 + c] = hi;
            *(u32x2*)&sAlo[buf][r * 40 + c] = lo;
        }
        #pragma unroll
        for (int j = 0; j < 2; j++) {           // B tile: 64x32 fp32 = 512 float4
            int f = tid + 256 * j;
            int r = f >> 3, c = (f & 7) << 2;
            f32x4 b = *(const f32x4*)(B + (size_t)(n0 + r) * ldb + (k0 + c));
            u32x2 hi, lo; pack_hi_lo(b, hi, lo);
            *(u32x2*)&sBhi[buf][r * 40 + c] = hi;
            *(u32x2*)&sBlo[buf][r * 40 + c] = lo;
        }
    };

    stage(0, 0);
    const int nk = K >> 5;
    for (int kc = 0; kc < nk; kc++) {
        __syncthreads();
        if (kc + 1 < nk) stage((kc + 1) & 1, (kc + 1) << 5);
        const int buf = kc & 1;
        const unsigned short* arh = &sAhi[buf][(wave * 16 + lr) * 40];
        const unsigned short* arl = &sAlo[buf][(wave * 16 + lr) * 40];
        v16bf ahi = frag_a(arh, lg);
        v16bf alo = frag_a(arl, lg);
        #pragma unroll
        for (int nt = 0; nt < 4; nt++) {
            const unsigned short* brh = &sBhi[buf][(nt * 16 + lr) * 40];
            const unsigned short* brl = &sBlo[buf][(nt * 16 + lr) * 40];
            v16bf bhi = frag_b(brh, lg);
            v16bf blo = frag_b(brl, lg);
            acc[nt] = __builtin_amdgcn_wmma_f32_16x16x32_bf16(false, ahi, false, bhi, (short)0, acc[nt], false, false);
            acc[nt] = __builtin_amdgcn_wmma_f32_16x16x32_bf16(false, ahi, false, blo, (short)0, acc[nt], false, false);
            acc[nt] = __builtin_amdgcn_wmma_f32_16x16x32_bf16(false, alo, false, bhi, (short)0, acc[nt], false, false);
        }
    }

    #pragma unroll
    for (int nt = 0; nt < 4; nt++) {
        const int col = n0 + nt * 16 + lr;
        #pragma unroll
        for (int v = 0; v < 8; v++) {
            const int row = m0 + wave * 16 + v + 8 * lg;
            C[(size_t)row * ldc + col] = acc[nt][v];
        }
    }
}

// ---------------------------------------------------------------------------
// In-place RoPE on Q (32 heads) and K (8 heads) inside the qkv buffer.
// One thread per (t, head, i<64) rotation pair.
// ---------------------------------------------------------------------------
__global__ __launch_bounds__(256) void rope_kernel(float* __restrict__ qkv,
                                                   const int* __restrict__ positions)
{
    const int idx  = blockIdx.x * 256 + threadIdx.x;
    const int i    = idx & 63;
    const int head = (idx >> 6) % 40;
    const int t    = idx / (64 * 40);
    const float pos = (float)positions[t];
    const float inv = __expf(-(float)i * (13.815510557964274f / 64.f)); // ln(1e6)/64
    float s, c;
    __sincosf(pos * inv, &s, &c);
    const int col = (head < HQ_) ? head * D_ : HQ_ * D_ + (head - HQ_) * D_;
    float* p = qkv + (size_t)t * QKV_N + col + i;
    const float x1 = p[0], x2 = p[64];
    p[0]  = x1 * c - x2 * s;
    p[64] = x2 * c + x1 * s;
}

// ---------------------------------------------------------------------------
// Sliding-window attention with sink, flash-style online softmax, fp32 WMMA.
// Workgroup = (kv head, 64-query block); wave = (q-head g, 32-query half).
// ---------------------------------------------------------------------------
__global__ __launch_bounds__(256) void swa_sink_kernel(
    const float* __restrict__ qkv, const float* __restrict__ sink_bias,
    float* __restrict__ attn)
{
    __shared__ float sK[16 * 132];        // 16 keys x 128, pitch-padded
    __shared__ float sV[16 * 100];        // 16 keys x 96, pitch-padded
    __shared__ float sP[8 * 2 * 16 * 20]; // per-wave P transpose scratch

    const int tid  = threadIdx.x;
    const int wave = tid >> 5, lane = tid & 31;
    const int lg   = lane >> 4, lr = lane & 15;
    const int qb   = blockIdx.x;          // 0..63
    const int kh   = blockIdx.y;          // 0..7
    const int g    = wave & 3;
    const int mh   = wave >> 2;
    const int h    = kh * G_ + g;
    const int qstart = qb * 64 + mh * 32;

    // Q fragments: 2 rowtiles x 32 K-chunks of f32 16x16x4 A-layout
    v2f qf[2][32];
    {
        const float* q0 = qkv + (size_t)(qstart + lr) * QKV_N + h * D_ + 2 * lg;
        const float* q1 = qkv + (size_t)(qstart + 16 + lr) * QKV_N + h * D_ + 2 * lg;
        #pragma unroll
        for (int c = 0; c < 32; c++) {
            qf[0][c] = *(const v2f*)(q0 + c * 4);
            qf[1][c] = *(const v2f*)(q1 + c * 4);
        }
    }

    const v8f vzero = {0.f, 0.f, 0.f, 0.f, 0.f, 0.f, 0.f, 0.f};
    v8f   accO[2][6];
    float Mst[2][8], Lst[2][8];
    #pragma unroll
    for (int rt = 0; rt < 2; rt++) {
        #pragma unroll
        for (int nt = 0; nt < 6; nt++) accO[rt][nt] = vzero;
        #pragma unroll
        for (int v = 0; v < 8; v++) { Mst[rt][v] = -1e30f; Lst[rt][v] = 0.f; }
    }

    const float scale = 0.08838834764831845f; // 1/sqrt(128)
    const float sink  = sink_bias[h];
    const int t0 = (qb * 4 - 32 > 0) ? (qb * 4 - 32) : 0;
    const int t1 = qb * 4 + 3;
    float* myP0 = &sP[(wave * 2 + 0) * 320];
    float* myP1 = &sP[(wave * 2 + 1) * 320];

    for (int kt = t0; kt <= t1; kt++) {
        __syncthreads();
        #pragma unroll
        for (int j = 0; j < 2; j++) {          // K tile: 512 float4
            int f = tid + 256 * j;
            int kr = f >> 5, kc = (f & 31) << 2;
            f32x4 kv = *(const f32x4*)(qkv + (size_t)(kt * 16 + kr) * QKV_N + (HQ_ * D_ + kh * D_) + kc);
            *(f32x4*)&sK[kr * 132 + kc] = kv;
        }
        #pragma unroll
        for (int j = 0; j < 2; j++) {          // V tile: 384 float4
            int f = tid + 256 * j;
            if (f < 384) {
                int vr = f / 24, vc = (f % 24) << 2;
                f32x4 vv = *(const f32x4*)(qkv + (size_t)(kt * 16 + vr) * QKV_N + (HQ_ * D_ + HKV_ * D_ + kh * DV_) + vc);
                *(f32x4*)&sV[vr * 100 + vc] = vv;
            }
        }
        __syncthreads();

        const int kbase = kt * 16;
        const int key   = kbase + lr;          // this lane's key column
        if (kbase <= qstart + 31 && kbase + 15 >= qstart - (W_ - 1)) { // wave-uniform
            #pragma unroll
            for (int rt = 0; rt < 2; rt++) {
                // S = Q * K^T over D=128: 32 chained f32 WMMAs
                v8f S = vzero;
                #pragma unroll
                for (int c = 0; c < 32; c++) {
                    v2f kf = *(const v2f*)&sK[lr * 132 + c * 4 + 2 * lg];
                    S = __builtin_amdgcn_wmma_f32_16x16x4_f32(false, qf[rt][c], false, kf, (short)0, S, false, false);
                }
                // online softmax: C-layout row = v + 8*(lane/16); 16-lane row reductions
                float fct[8], Pv[8];
                #pragma unroll
                for (int v = 0; v < 8; v++) {
                    const int row = qstart + rt * 16 + v + 8 * lg;
                    float s = S[v] * scale;
                    const bool valid = (key <= row) && (key >= row - (W_ - 1));
                    s = valid ? s : -1e30f;
                    float m = s;
                    m = fmaxf(m, __shfl_xor(m, 1));
                    m = fmaxf(m, __shfl_xor(m, 2));
                    m = fmaxf(m, __shfl_xor(m, 4));
                    m = fmaxf(m, __shfl_xor(m, 8));
                    const float Mold = Mst[rt][v];
                    const float Mnew = fmaxf(Mold, m);
                    float p = (s > -1e29f) ? __expf(s - Mnew) : 0.f;
                    float rs = p;
                    rs += __shfl_xor(rs, 1);
                    rs += __shfl_xor(rs, 2);
                    rs += __shfl_xor(rs, 4);
                    rs += __shfl_xor(rs, 8);
                    const float f0 = __expf(Mold - Mnew);
                    Lst[rt][v] = Lst[rt][v] * f0 + rs;
                    Mst[rt][v] = Mnew;
                    fct[v] = f0;
                    Pv[v]  = p;
                }
                #pragma unroll
                for (int nt = 0; nt < 6; nt++)
                    #pragma unroll
                    for (int v = 0; v < 8; v++)
                        accO[rt][nt][v] *= fct[v];
                // transpose P (C-layout -> A-fragment) through per-wave LDS scratch
                float* ps = rt ? myP1 : myP0;
                #pragma unroll
                for (int v = 0; v < 8; v++) ps[(v + 8 * lg) * 20 + lr] = Pv[v];
                v2f pf[4];
                #pragma unroll
                for (int c = 0; c < 4; c++) pf[c] = *(const v2f*)&ps[lr * 20 + c * 4 + 2 * lg];
                // O += P * V (K=16 in 4 chunks, 6 column tiles of 16)
                #pragma unroll
                for (int nt = 0; nt < 6; nt++) {
                    v8f o = accO[rt][nt];
                    #pragma unroll
                    for (int c = 0; c < 4; c++) {
                        v2f vf;
                        vf.x = sV[(c * 4 + 2 * lg) * 100 + nt * 16 + lr];
                        vf.y = sV[(c * 4 + 2 * lg + 1) * 100 + nt * 16 + lr];
                        o = __builtin_amdgcn_wmma_f32_16x16x4_f32(false, pf[c], false, vf, (short)0, o, false, false);
                    }
                    accO[rt][nt] = o;
                }
            }
        }
    }

    // epilogue: divide by (L + exp(sink - M)) and store
    #pragma unroll
    for (int rt = 0; rt < 2; rt++) {
        float inv[8];
        #pragma unroll
        for (int v = 0; v < 8; v++)
            inv[v] = 1.f / (Lst[rt][v] + __expf(sink - Mst[rt][v]));
        #pragma unroll
        for (int nt = 0; nt < 6; nt++) {
            const int col = h * DV_ + nt * 16 + lr;
            #pragma unroll
            for (int v = 0; v < 8; v++) {
                const int row = qstart + rt * 16 + v + 8 * lg;
                attn[(size_t)row * ATT_N + col] = accO[rt][nt][v] * inv[v];
            }
        }
    }
}

extern "C" void kernel_launch(void* const* d_in, const int* in_sizes, int n_in,
                              void* d_out, int out_size, void* d_ws, size_t ws_size,
                              hipStream_t stream) {
    const int*   positions = (const int*)d_in[0];
    const float* hidden    = (const float*)d_in[1];
    const float* qkv_w     = (const float*)d_in[2];   // (5888, 2048) row-major
    const float* o_w       = (const float*)d_in[3];   // (2048, 3072) row-major
    const float* sinkb     = (const float*)d_in[4];   // (32,)
    float* out = (float*)d_out;                       // (4096, 2048)

    float* ws   = (float*)d_ws;
    float* qkv  = ws;                                 // (4096, 5888)
    float* attn = ws + (size_t)T_ * QKV_N;            // (4096, 3072)

    // 1) QKV = hidden @ qkv_w^T
    gemm_f32_bf16split<<<dim3(QKV_N / 64, T_ / 128), 256, 0, stream>>>(
        hidden, HID_, qkv_w, HID_, qkv, QKV_N, HID_);
    // 2) RoPE on Q and K in place
    rope_kernel<<<dim3((T_ * 40 * 64) / 256), 256, 0, stream>>>(qkv, positions);
    // 3) sliding-window attention with sink
    swa_sink_kernel<<<dim3(T_ / 64, HKV_), 256, 0, stream>>>(qkv, sinkb, attn);
    // 4) out = attn @ o_w^T
    gemm_f32_bf16split<<<dim3(HID_ / 64, T_ / 128), 256, 0, stream>>>(
        attn, ATT_N, o_w, ATT_N, out, HID_, ATT_N);
}
